// PlacementCNN_10960756540056
// MI455X (gfx1250) — matrix operand order repaired and checked by reading, not compiled, not run on hardware
//
#include <hip/hip_runtime.h>
#include <hip/hip_bf16.h>
#include <math.h>

// ---------------------------------------------------------------------------
// PlacementCNN for MI455X (gfx1250, wave32, WMMA)
//   conv0(7x3)+relu+pool3  -> p0   (T+8, 10, 26)  f16   [VALU kernel]
//   conv1(3x3)+relu+pool3  -> feats(T+6, 160)     f16   [VALU kernel]
//   GEMM1 h0 = win @ Wx.T  -> WMMA f16 16x16x32, K=1120, rolled k-loop
//     (A/B advance +32 halves per step), wave tile 64x64 (4x4 WMMA tiles),
//     16 loads feed 16 WMMAs per step, all tile offsets in load immediates.
//   raw h0 tile stored once to LDS (128x256 f16); per diff d: block builds
//   f16 bias row (b0 + W0[:,1120+dc]) in LDS, GEMM2 A-frags apply
//   relu(h0+bias) on the fly -> WMMA vs W1 -> relu . Wout -> shfl reduce
//   -> sigmoid.
// ---------------------------------------------------------------------------

typedef __attribute__((ext_vector_type(16))) _Float16 v16h;
typedef __attribute__((ext_vector_type(8)))  _Float16 v8h;
typedef __attribute__((ext_vector_type(8)))  float    v8f;

#define T_LEN 65536
#define R1 7
#define NDIFF 5
#define TBLK 128   // time rows per block (2x4 wave grid of 64x64 tiles)

// ------------------------- weight conversion f32 -> f16 --------------------
__global__ void convert_weights_kernel(const float* __restrict__ W0,
                                       const float* __restrict__ W1,
                                       _Float16* __restrict__ Wxh,   // 256x1120
                                       _Float16* __restrict__ W1h) { // 128x256
  int i = blockIdx.x * blockDim.x + threadIdx.x;
  if (i < 256 * 1120) {
    int r = i / 1120, c = i % 1120;
    Wxh[i] = (_Float16)W0[r * 1125 + c];
  }
  if (i < 128 * 256) {
    W1h[i] = (_Float16)W1[i];
  }
}

// ------------------------- conv0 + relu + pool3 -----------------------------
// x: (T,80,3); out p0: [r][oc][q] r in [0,T+8), oc in [0,10), q in [0,26)
__global__ void conv0_pool_kernel(const float* __restrict__ x,
                                  const float* __restrict__ w,   // (10,3,7,3)
                                  const float* __restrict__ b,
                                  _Float16* __restrict__ p0) {
  int idx = blockIdx.x * blockDim.x + threadIdx.x;
  const int NOUT = (T_LEN + 8) * 10 * 26;
  if (idx >= NOUT) return;
  int q  = idx % 26;
  int rc = idx / 26;
  int oc = rc % 10;
  int r  = rc / 10;

  float m3 = -3.4e38f;
#pragma unroll
  for (int pp = 0; pp < 3; ++pp) {
    int fo = q * 3 + pp;                 // 0..77
    float s = b[oc];
#pragma unroll
    for (int c = 0; c < 3; ++c) {
#pragma unroll
      for (int kh = 0; kh < 7; ++kh) {
        int ti = r + kh - R1;            // padded time index
        if (ti >= 0 && ti < T_LEN) {
          const float* xr = x + ((size_t)ti * 80 + fo) * 3 + c;
          const float* wr = w + ((oc * 3 + c) * 7 + kh) * 3;
          s += xr[0] * wr[0] + xr[3] * wr[1] + xr[6] * wr[2];
        }
      }
    }
    m3 = fmaxf(m3, s);
  }
  p0[idx] = (_Float16)fmaxf(m3, 0.f);
}

// ------------------------- conv1 + relu + pool3 -----------------------------
// p0: [r][c][q26]; feats: [r2][q2*20 + oc], r2 in [0,T+6), q2 in [0,8), oc in [0,20)
__global__ void conv1_pool_kernel(const _Float16* __restrict__ p0,
                                  const float* __restrict__ w,   // (20,10,3,3)
                                  const float* __restrict__ b,
                                  _Float16* __restrict__ feats) {
  int idx = blockIdx.x * blockDim.x + threadIdx.x;
  const int NOUT = (T_LEN + 6) * 160;
  if (idx >= NOUT) return;
  int rem = idx % 160;
  int r2  = idx / 160;
  int oc  = rem % 20;
  int q2  = rem / 20;

  float m3 = -3.4e38f;
#pragma unroll
  for (int pp = 0; pp < 3; ++pp) {
    int fo = q2 * 3 + pp;                // 0..23
    float s = b[oc];
#pragma unroll
    for (int c = 0; c < 10; ++c) {
#pragma unroll
      for (int kh = 0; kh < 3; ++kh) {
        const _Float16* pr = p0 + (size_t)(r2 + kh) * 260 + c * 26 + fo;
        const float*    wr = w + ((oc * 10 + c) * 3 + kh) * 3;
        s += (float)pr[0] * wr[0] + (float)pr[1] * wr[1] + (float)pr[2] * wr[2];
      }
    }
    m3 = fmaxf(m3, s);
  }
  feats[idx] = (_Float16)fmaxf(m3, 0.f);
}

// ------------------------- fused WMMA MLP -----------------------------------
__global__ __launch_bounds__(256)
void fused_mlp_kernel(const _Float16* __restrict__ feats,  // (T+6,160)
                      const _Float16* __restrict__ Wxh,    // (256,1120)
                      const _Float16* __restrict__ W1h,    // (128,256)
                      const float* __restrict__ W0,        // (256,1125) for diff bias
                      const float* __restrict__ b0,
                      const float* __restrict__ b1,
                      const float* __restrict__ Wout,      // (128)
                      const float* __restrict__ bout,
                      const int*   __restrict__ diffs,
                      float* __restrict__ out) {           // (NDIFF, T)
  __shared__ _Float16 ldsH[TBLK * 256];                    // 64 KB raw h0 tile
  __shared__ _Float16 biasLds[256];                        // per-d bias row

  const int lane   = threadIdx.x & 31;
  const int wave   = threadIdx.x >> 5;
  const int mlane  = lane & 15;
  const int hiHalf = lane >> 4;            // 0: lanes 0-15, 1: lanes 16-31
  const int kbaseA = hiHalf * 16;          // A-frag K base (halves)
  const int koffB  = hiHalf * 16;          // B-frag K base (halves)

  const int wm = wave & 1;                 // M quadrant (2 x 64 rows)
  const int wn = wave >> 1;                // N quadrant (4 x 64 cols)
  const int rowBase = blockIdx.x * TBLK + wm * 64;   // wave's first time row
  const int colBase = wn * 64;                       // wave's first h0 col

  const v8f vzero = {0.f, 0.f, 0.f, 0.f, 0.f, 0.f, 0.f, 0.f};

  // ---- GEMM1: 64x64 wave tile, rolled k-loop (35 steps of K=32) -----------
  v8f acc[4][4];
#pragma unroll
  for (int i = 0; i < 4; ++i)
#pragma unroll
    for (int j = 0; j < 4; ++j) acc[i][j] = vzero;

  // A and B streams both advance exactly +32 halves per k-step:
  //   A: (tap*160 + c5*32) == ks*32   (160 = 5*32),  B: kg == ks*32.
  const _Float16* aPtr = feats + (size_t)(rowBase + mlane) * 160 + kbaseA;
  const _Float16* bPtr = Wxh   + (size_t)(colBase + mlane) * 1120 + koffB;

#pragma unroll 1
  for (int ks = 0; ks < 35; ++ks) {
    v16h afrag[4];
#pragma unroll
    for (int mt = 0; mt < 4; ++mt) {
      const _Float16* pa = aPtr + mt * (16 * 160);   // +5120 B per M-tile
      v8h alo = *(const v8h*)(pa);
      v8h ahi = *(const v8h*)(pa + 8);
      afrag[mt] = __builtin_shufflevector(alo, ahi,
              0, 1, 2, 3, 4, 5, 6, 7, 8, 9, 10, 11, 12, 13, 14, 15);
    }
    v16h bfrag[4];
#pragma unroll
    for (int ntq = 0; ntq < 4; ++ntq)
      bfrag[ntq] = *(const v16h*)(bPtr + ntq * (16 * 1120));  // +35840 B
#pragma unroll
    for (int mt = 0; mt < 4; ++mt)
#pragma unroll
      for (int ntq = 0; ntq < 4; ++ntq)
        acc[mt][ntq] = __builtin_amdgcn_wmma_f32_16x16x32_f16(
            false, afrag[mt], false, bfrag[ntq], (short)0, acc[mt][ntq],
            false, false);
    aPtr += 32;
    bPtr += 32;
  }

  // ---- store raw h0 (f16, no bias) once; frees all 128 acc VGPRs ----------
  {
    _Float16* stBase = ldsH + (size_t)(wm * 64 + hiHalf * 8) * 256
                            + colBase + mlane;
#pragma unroll
    for (int ntq = 0; ntq < 4; ++ntq)
#pragma unroll
      for (int mt = 0; mt < 4; ++mt)
#pragma unroll
        for (int v = 0; v < 8; ++v)
          stBase[(mt * 16 + v) * 256 + ntq * 16] =
              (_Float16)acc[mt][ntq][v];
  }
  __syncthreads();

  // ---- per-lane invariant epilogue constants ------------------------------
  float b1l[8], wol[8];
#pragma unroll
  for (int nt = 0; nt < 8; ++nt) {
    int n = nt * 16 + mlane;
    b1l[nt] = b1[n];
    wol[nt] = Wout[n];
  }
  const float bo = bout[0];
  const int tid = threadIdx.x;

  const _Float16* a2Base = ldsH + (size_t)wave * 16 * 256
                                + (size_t)mlane * 256 + kbaseA;
  const _Float16* bsBase = biasLds + kbaseA;
  const _Float16* b2Base = W1h + (size_t)mlane * 256 + koffB;

  // ---- per-difficulty: bias row -> GEMM2(relu(h0+bias)) -> logits ---------
  for (int d = 0; d < NDIFF; ++d) {
    const int dc = diffs[d];
    // block-cooperative f16 bias row: b0[col] + W0[col,1120+dc]
    biasLds[tid] = (_Float16)(b0[tid] + W0[(size_t)tid * 1125 + 1120 + dc]);
    __syncthreads();

    v8f acc2[8];
#pragma unroll
    for (int i = 0; i < 8; ++i) acc2[i] = vzero;

    const _Float16* a2Ptr = a2Base;
    const _Float16* bsPtr = bsBase;
    const _Float16* b2Ptr = b2Base;
#pragma unroll 1
    for (int ks = 0; ks < 8; ++ks) {
      v8h hlo = *(const v8h*)(a2Ptr);
      v8h hhi = *(const v8h*)(a2Ptr + 8);
      v8h slo = *(const v8h*)(bsPtr);
      v8h shi = *(const v8h*)(bsPtr + 8);
      v16h hsum = __builtin_shufflevector(hlo, hhi,
              0, 1, 2, 3, 4, 5, 6, 7, 8, 9, 10, 11, 12, 13, 14, 15);
      v16h bsum = __builtin_shufflevector(slo, shi,
              0, 1, 2, 3, 4, 5, 6, 7, 8, 9, 10, 11, 12, 13, 14, 15);
      v16h a2;
#pragma unroll
      for (int j = 0; j < 16; ++j) {
        _Float16 hv = (_Float16)(hsum[j] + bsum[j]);
        a2[j] = hv > (_Float16)0 ? hv : (_Float16)0;
      }
#pragma unroll
      for (int nt = 0; nt < 8; ++nt) {
        v16h b2 = *(const v16h*)(b2Ptr + nt * (16 * 256));   // +8192 B
        acc2[nt] = __builtin_amdgcn_wmma_f32_16x16x32_f16(
            false, a2, false, b2, (short)0, acc2[nt], false, false);
      }
      a2Ptr += 32;
      bsPtr += 32;
      b2Ptr += 32;
    }

    // logits: relu(a1 + b1) . Wout, reduce over N via 16-lane halves
#pragma unroll
    for (int v = 0; v < 8; ++v) {
      float s = 0.f;
#pragma unroll
      for (int nt = 0; nt < 8; ++nt) {
        float av = acc2[nt][v] + b1l[nt];
        av = av > 0.f ? av : 0.f;
        s += av * wol[nt];
      }
      s += __shfl_xor(s, 1);
      s += __shfl_xor(s, 2);
      s += __shfl_xor(s, 4);
      s += __shfl_xor(s, 8);
      if (mlane == 0) {
        int m = v + hiHalf * 8;
        float lg = s + bo;
        out[(size_t)d * T_LEN + blockIdx.x * TBLK + wave * 16 + m] =
            1.f / (1.f + __expf(-lg));
      }
    }
    __syncthreads();   // biasLds reused next d
  }
}

// ---------------------------------------------------------------------------
extern "C" void kernel_launch(void* const* d_in, const int* in_sizes, int n_in,
                              void* d_out, int out_size, void* d_ws, size_t ws_size,
                              hipStream_t stream) {
  const float* x     = (const float*)d_in[0];
  const int*   diffs = (const int*)d_in[1];
  const float* c0w   = (const float*)d_in[2];
  const float* c0b   = (const float*)d_in[3];
  const float* c1w   = (const float*)d_in[4];
  const float* c1b   = (const float*)d_in[5];
  const float* W0    = (const float*)d_in[6];
  const float* b0    = (const float*)d_in[7];
  const float* W1    = (const float*)d_in[8];
  const float* b1    = (const float*)d_in[9];
  const float* Wout  = (const float*)d_in[10];
  const float* bout  = (const float*)d_in[11];
  float* out = (float*)d_out;

  // workspace layout (bytes, all offsets 32B-aligned)
  char* ws = (char*)d_ws;
  _Float16* Wxh   = (_Float16*)(ws);                       // 256*1120*2 = 573440
  _Float16* W1h   = (_Float16*)(ws + 573440);              // 128*256*2  = 65536
  _Float16* p0    = (_Float16*)(ws + 638976);              // (T+8)*260*2 = 34082880
  _Float16* feats = (_Float16*)(ws + 34721856);            // (T+6)*160*2 = 20973440

  {
    int n = 256 * 1120;
    convert_weights_kernel<<<(n + 255) / 256, 256, 0, stream>>>(W0, W1, Wxh, W1h);
  }
  {
    int n = (T_LEN + 8) * 10 * 26;
    conv0_pool_kernel<<<(n + 255) / 256, 256, 0, stream>>>(x, c0w, c0b, p0);
  }
  {
    int n = (T_LEN + 6) * 160;
    conv1_pool_kernel<<<(n + 255) / 256, 256, 0, stream>>>(p0, c1w, c1b, feats);
  }
  {
    fused_mlp_kernel<<<T_LEN / TBLK, 256, 0, stream>>>(
        feats, Wxh, W1h, W0, b0, b1, Wout, bout, diffs, out);
  }
}
